// VanillaRNN_58686433132698
// MI455X (gfx1250) — compile-verified, matrix operand order checked by reading, not run on hardware
//
#include <hip/hip_runtime.h>
#include <hip/hip_bf16.h>

// Sizes from the reference
#define RT 256   // timesteps
#define RB 256   // batch
#define RD 128   // input dim
#define RH 1024  // hidden dim
#define RC 10    // classes

typedef __attribute__((ext_vector_type(16))) _Float16 v16h;
typedef __attribute__((ext_vector_type(8)))  float    v8f;

union Frag16 {
    v16h h;
    uint4 u[2];
};

// ---------------------------------------------------------------------------
// f32 -> f16 conversion (grid-stride)
// ---------------------------------------------------------------------------
__global__ void rnn_cvt_f32_to_f16(const float* __restrict__ in,
                                   _Float16* __restrict__ out, int n) {
    int i = blockIdx.x * blockDim.x + threadIdx.x;
    int stride = gridDim.x * blockDim.x;
    for (; i < n; i += stride) out[i] = (_Float16)in[i];
}

__global__ void rnn_zero_f16(_Float16* __restrict__ out, int n) {
    int i = blockIdx.x * blockDim.x + threadIdx.x;
    int stride = gridDim.x * blockDim.x;
    for (; i < n; i += stride) out[i] = (_Float16)0.0f;
}

// ---------------------------------------------------------------------------
// One RNN timestep:
//   h_out = tanh( x_t @ W_hx^T + h_in @ W_hh^T + b_h )     [B=256, H=1024]
//
// Register-blocked tiling: each wave owns a 32x32 output tile = 4 WMMA
// sub-tiles with 4 independent f32 accumulators. Per K-step: 2 A-frags +
// 2 B-frags feed 4 v_wmma_f32_16x16x32_f16 (1 fragment load per WMMA, and
// 4 independent accumulator chains for XDL ILP).
//
// 8 m-supertiles (B/32) x 32 n-supertiles (H/32) = 256 waves
//   -> 32 blocks x 8 waves; the 8 waves of a block share one n-supertile
//      so W fragments are reused 8-way through L0.
//
// Fragment layout (ISA 7.12.2, 16-bit, wave32):
//   A 16x32: r=lane&15 -> row M=r; hi=lane>>4; VGPR0..3 = K[hi*8 .. +7],
//            VGPR4..7 = K[16+hi*8 .. +7]  -> two b128 loads per fragment.
//   B 32x16: same pattern, lane&15 selecting column N (row of W, since
//            B[k][n] = W[n][k] and W is row-major).
//   C/D f32: lane&15 = N, VGPR v -> M = v + 8*hi.
// ---------------------------------------------------------------------------
__global__ __launch_bounds__(256) void rnn_step_kernel(
        const _Float16* __restrict__ x_t,    // [B, D]  f16
        const _Float16* __restrict__ w_hx,   // [H, D]  f16
        const _Float16* __restrict__ w_hh,   // [H, H]  f16
        const float*    __restrict__ b_h,    // [H]     f32
        const _Float16* __restrict__ h_in,   // [B, H]  f16
        _Float16*       __restrict__ h_out)  // [B, H]  f16
{
    const int wave = threadIdx.x >> 5;
    const int lane = threadIdx.x & 31;
    const int tile = blockIdx.x * 8 + wave;          // 0..255
    const int m_super = tile & 7;                    // 8 m-supertiles (B/32)
    const int n_super = tile >> 3;                   // 32 n-supertiles (H/32)
    const int base_m = m_super * 32;
    const int base_n = n_super * 32;

    const int r  = lane & 15;
    const int hi = lane >> 4;                        // 0 or 1

    // Per-lane fragment base pointers (offset hi*8 halves = 16B, aligned)
    const _Float16* pax0 = x_t  + (size_t)(base_m + r) * RD + hi * 8;
    const _Float16* pax1 = pax0 + (size_t)16 * RD;
    const _Float16* pbx0 = w_hx + (size_t)(base_n + r) * RD + hi * 8;
    const _Float16* pbx1 = pbx0 + (size_t)16 * RD;

    const _Float16* pah0 = h_in + (size_t)(base_m + r) * RH + hi * 8;
    const _Float16* pah1 = pah0 + (size_t)16 * RH;
    const _Float16* pbh0 = w_hh + (size_t)(base_n + r) * RH + hi * 8;
    const _Float16* pbh1 = pbh0 + (size_t)16 * RH;

    v8f acc00 = {};   // (m0, n0)
    v8f acc01 = {};   // (m0, n1)
    v8f acc10 = {};   // (m1, n0)
    v8f acc11 = {};   // (m1, n1)

    // ---- input projection: K over D = 128 (4 steps x 4 WMMAs) ----
#pragma unroll
    for (int k = 0; k < RD; k += 32) {
        Frag16 a0, a1, b0, b1;
        a0.u[0] = *(const uint4*)(pax0 + k);
        a0.u[1] = *(const uint4*)(pax0 + k + 16);
        a1.u[0] = *(const uint4*)(pax1 + k);
        a1.u[1] = *(const uint4*)(pax1 + k + 16);
        b0.u[0] = *(const uint4*)(pbx0 + k);
        b0.u[1] = *(const uint4*)(pbx0 + k + 16);
        b1.u[0] = *(const uint4*)(pbx1 + k);
        b1.u[1] = *(const uint4*)(pbx1 + k + 16);
        acc00 = __builtin_amdgcn_wmma_f32_16x16x32_f16(
                    false, a0.h, false, b0.h, (short)0, acc00, false, false);
        acc01 = __builtin_amdgcn_wmma_f32_16x16x32_f16(
                    false, a0.h, false, b1.h, (short)0, acc01, false, false);
        acc10 = __builtin_amdgcn_wmma_f32_16x16x32_f16(
                    false, a1.h, false, b0.h, (short)0, acc10, false, false);
        acc11 = __builtin_amdgcn_wmma_f32_16x16x32_f16(
                    false, a1.h, false, b1.h, (short)0, acc11, false, false);
    }

    // ---- recurrence: K over H = 1024 (32 steps x 4 WMMAs) ----
#pragma unroll 4
    for (int k = 0; k < RH; k += 32) {
        Frag16 a0, a1, b0, b1;
        a0.u[0] = *(const uint4*)(pah0 + k);
        a0.u[1] = *(const uint4*)(pah0 + k + 16);
        a1.u[0] = *(const uint4*)(pah1 + k);
        a1.u[1] = *(const uint4*)(pah1 + k + 16);
        b0.u[0] = *(const uint4*)(pbh0 + k);
        b0.u[1] = *(const uint4*)(pbh0 + k + 16);
        b1.u[0] = *(const uint4*)(pbh1 + k);
        b1.u[1] = *(const uint4*)(pbh1 + k + 16);
        acc00 = __builtin_amdgcn_wmma_f32_16x16x32_f16(
                    false, a0.h, false, b0.h, (short)0, acc00, false, false);
        acc01 = __builtin_amdgcn_wmma_f32_16x16x32_f16(
                    false, a0.h, false, b1.h, (short)0, acc01, false, false);
        acc10 = __builtin_amdgcn_wmma_f32_16x16x32_f16(
                    false, a1.h, false, b0.h, (short)0, acc10, false, false);
        acc11 = __builtin_amdgcn_wmma_f32_16x16x32_f16(
                    false, a1.h, false, b1.h, (short)0, acc11, false, false);
    }

    // ---- bias + tanh + store (C/D layout: N = lane&15, M = v + 8*hi) ----
    const int n0 = base_n + r;
    const int n1 = n0 + 16;
    const float bias0 = b_h[n0];
    const float bias1 = b_h[n1];
#pragma unroll
    for (int v = 0; v < 8; ++v) {
        const int m0 = base_m + v + hi * 8;
        const int m1 = m0 + 16;
        h_out[(size_t)m0 * RH + n0] = (_Float16)tanhf(acc00[v] + bias0);
        h_out[(size_t)m0 * RH + n1] = (_Float16)tanhf(acc01[v] + bias1);
        h_out[(size_t)m1 * RH + n0] = (_Float16)tanhf(acc10[v] + bias0);
        h_out[(size_t)m1 * RH + n1] = (_Float16)tanhf(acc11[v] + bias1);
    }
}

// ---------------------------------------------------------------------------
// Readout: p[b][c] = sum_h hT[b][h] * w_ph[c][h] + b_p[c]   (B=256, C=10)
// ---------------------------------------------------------------------------
__global__ void rnn_readout_kernel(const _Float16* __restrict__ hT,  // [B,H]
                                   const float* __restrict__ w_ph,   // [C,H]
                                   const float* __restrict__ b_p,    // [C]
                                   float* __restrict__ out)          // [B,C]
{
    const int c = blockIdx.x;    // 0..C-1
    const int b = threadIdx.x;   // 0..B-1
    const _Float16* hr = hT + (size_t)b * RH;
    const float*    wr = w_ph + (size_t)c * RH;
    float s = 0.0f;
#pragma unroll 4
    for (int k = 0; k < RH; ++k) s += (float)hr[k] * wr[k];
    out[(size_t)b * RC + c] = s + b_p[c];
}

// ---------------------------------------------------------------------------
// Launch
// ---------------------------------------------------------------------------
extern "C" void kernel_launch(void* const* d_in, const int* in_sizes, int n_in,
                              void* d_out, int out_size, void* d_ws, size_t ws_size,
                              hipStream_t stream) {
    const float* x    = (const float*)d_in[0];  // [T,B,D]
    const float* w_hx = (const float*)d_in[1];  // [H,D]
    const float* w_hh = (const float*)d_in[2];  // [H,H]
    const float* w_ph = (const float*)d_in[3];  // [C,H]
    const float* b_h  = (const float*)d_in[4];  // [H]
    const float* b_p  = (const float*)d_in[5];  // [C]
    float* out = (float*)d_out;                 // [B,C]

    // Workspace layout (halves). Total ~19.3 MB.
    _Float16* ws    = (_Float16*)d_ws;
    _Float16* x16   = ws;                                         // T*B*D
    _Float16* whx16 = x16   + (size_t)RT * RB * RD;               // H*D
    _Float16* whh16 = whx16 + (size_t)RH * RD;                    // H*H
    _Float16* hbuf0 = whh16 + (size_t)RH * RH;                    // B*H
    _Float16* hbuf1 = hbuf0 + (size_t)RB * RH;                    // B*H

    const int nX  = RT * RB * RD;   // 8,388,608
    const int nWx = RH * RD;        // 131,072
    const int nWh = RH * RH;        // 1,048,576
    const int nH  = RB * RH;        // 262,144

    rnn_cvt_f32_to_f16<<<4096, 256, 0, stream>>>(x,    x16,   nX);
    rnn_cvt_f32_to_f16<<<512,  256, 0, stream>>>(w_hx, whx16, nWx);
    rnn_cvt_f32_to_f16<<<2048, 256, 0, stream>>>(w_hh, whh16, nWh);
    rnn_zero_f16<<<512, 256, 0, stream>>>(hbuf0, nH);

    // Sequential recurrence: one launch per timestep (stream order = sync).
    for (int t = 0; t < RT; ++t) {
        const _Float16* hin  = (t & 1) ? hbuf1 : hbuf0;
        _Float16*       hout = (t & 1) ? hbuf0 : hbuf1;
        rnn_step_kernel<<<32, 256, 0, stream>>>(
            x16 + (size_t)t * RB * RD, whx16, whh16, b_h, hin, hout);
    }

    // t = 255 wrote hbuf0 -> final hidden state.
    rnn_readout_kernel<<<RC, RB, 0, stream>>>(hbuf0, w_ph, b_p, out);
}
